// tensor_42992622633046
// MI455X (gfx1250) — matrix operand order repaired and checked by reading, not compiled, not run on hardware
//
#include <hip/hip_runtime.h>

// ---------------------------------------------------------------------------
// CDNA5 (gfx1250) fused tensor-fusion MLP
//   y1 = relu(fusion @ W1^T + b1)   -- 36 GFLOP, W1 = 140.6 MB  (dominant)
//   y2 = relu(y1 @ W2^T + b2)
//   y3 = softmax(y2 @ W3^T + b3), output = (y3, y3)
// v4: fully branch-free hot loop -- A-generation uses predicated selects
//     (no EXEC-mask churn), carried indices (no divisions), LDS-cooperative
//     fragment staging, clause'd b128 W1 streaming + prefetch.
// ---------------------------------------------------------------------------

typedef __attribute__((ext_vector_type(16))) __bf16 v16bf;
typedef __attribute__((ext_vector_type(8)))  float  v8f;

#define IN_SIZE 64
#define D1      65
#define D2      4225       // 65*65
#define FUSED   274625     // 65^3
#define BATCH   512
#define HID     128
#define OUT_N   8

#define KSTEP    32
#define NSTEPS   8583      // ceil(FUSED/32)
#define NFULL    8582      // steps with all 32 K in-bounds (s <= 8581)
#define KCHUNKS  32
#define SPC      269       // ceil(NSTEPS/KCHUNKS)
#define MBLOCKS  8         // 64 rows per block

struct U256 { uint4 lo, hi; };   // 32-byte carrier for a v16bf fragment

__device__ __forceinline__ unsigned pk_bf16x2(float a, float b) {
  unsigned short ua = __builtin_bit_cast(unsigned short, (__bf16)a);
  unsigned short ub = __builtin_bit_cast(unsigned short, (__bf16)b);
  return (unsigned)ua | ((unsigned)ub << 16);
}

// ---------------- Kernel 1: fusion GEMM (K-split partial sums) -------------
__global__ __launch_bounds__(256) void fusion_gemm_k1(
    const float* __restrict__ x, const float* __restrict__ W1,
    float* __restrict__ slabs)
{
  __shared__ float    Xs[64 * 3 * D1];      // 48.75 KB: 64 rows x {ah,vh,lh}
  __shared__ unsigned Blds[8 * 32 * 8];     // 8 KB: 8 N-tile B fragments
  __shared__ unsigned Alds[4 * 32 * 8];     // 4 KB: 4 M-tile A fragments

  const int tid    = threadIdx.x;
  const int kchunk = blockIdx.x;
  const int mblock = blockIdx.y;
  const int row0   = mblock * 64;

  // Stage this M-block's x rows (with leading 1) into LDS.
  for (int u = tid; u < 64 * 3 * D1; u += 256) {
    int rl  = u / (3 * D1);
    int rem = u - rl * (3 * D1);
    int cc  = rem / D1;
    int idx = rem - cc * D1;
    Xs[u] = (idx == 0) ? 1.0f
          : x[((size_t)(row0 + rl) * 3 + cc) * IN_SIZE + (idx - 1)];
  }

  const int lane = tid & 31;
  const int w    = tid >> 5;      // wave 0..7 == its N-tile
  const int hi   = lane >> 4;
  const int m    = lane & 15;

  const int s0 = kchunk * SPC;
  int s1 = s0 + SPC; if (s1 > NFULL) s1 = NFULL;       // main loop: full steps

  // --- A-generation identity: one 8-element run per thread -----------------
  // unit = (fragment f 0..3, fragment-lane al 0..31, run ar 0..1)
  const int f  = tid >> 6;
  const int rm = tid & 63;
  const int al = rm & 31;
  const int ar = rm >> 5;
  const int arow_l = f * 16 + (al & 15);               // source row in Xs
  const float* gAH = &Xs[(arow_l * 3) * D1];
  const float* gVH = gAH + D1;
  const float* gLH = gAH + 2 * D1;
  const int aoff = ((al >> 4) * 16) + ar * 8;          // run-start k offset
  // master indices for kg = s0*32 + aoff (single divmod, then carried)
  int kg0 = s0 * KSTEP + aoff;
  int ii  = kg0 / D2;
  int rr  = kg0 - ii * D2;
  int jj  = rr / D1;
  int kk0 = rr - jj * D1;

  // --- B-staging identity: thread == (n-tile = w, lane) --------------------
  const size_t wrow = (size_t)((lane & 15) + w * 16) * FUSED;

  v8f c[4] = {};

  __syncthreads();
  for (int s = s0; s < s1; ++s) {
    const int k0 = s * KSTEP;

    // ---- stage B tile (32K x 128N), f32 -> bf16, fragment layout ----
    {
      const float* wp = W1 + wrow + (k0 + hi * 16);    // 16 contiguous f32
      __builtin_prefetch(wp + KSTEP, 0, 1);            // next K-step
      uint4 p0, p1;
      p0.x = pk_bf16x2(wp[0],  wp[1]);  p0.y = pk_bf16x2(wp[2],  wp[3]);
      p0.z = pk_bf16x2(wp[4],  wp[5]);  p0.w = pk_bf16x2(wp[6],  wp[7]);
      p1.x = pk_bf16x2(wp[8],  wp[9]);  p1.y = pk_bf16x2(wp[10], wp[11]);
      p1.z = pk_bf16x2(wp[12], wp[13]); p1.w = pk_bf16x2(wp[14], wp[15]);
      *(uint4*)&Blds[(unsigned)tid * 8]     = p0;
      *(uint4*)&Blds[(unsigned)tid * 8 + 4] = p1;
    }

    // ---- stage A fragments cooperatively, fully branch-free ----
    {
      // at most one j-boundary inside an 8-run: precompute next-j product
      int  j2 = jj + 1;
      bool jw = (j2 == D1);
      int  i2 = ii + (jw ? 1 : 0);
      j2 = jw ? 0 : j2;
      float p1v = gAH[ii] * gVH[jj];
      float p2v = gAH[i2] * gVH[j2];
      float v[8];
      #pragma unroll
      for (int e = 0; e < 8; ++e) {
        int  idx = kk0 + e;
        bool wr  = (idx >= D1);
        idx = wr ? idx - D1 : idx;
        v[e] = (wr ? p2v : p1v) * gLH[idx];
      }
      uint4 av;
      av.x = pk_bf16x2(v[0], v[1]); av.y = pk_bf16x2(v[2], v[3]);
      av.z = pk_bf16x2(v[4], v[5]); av.w = pk_bf16x2(v[6], v[7]);
      *(uint4*)&Alds[(unsigned)(f * 32 + al) * 8 + ar * 4] = av;
      // advance master indices by one K-step (32), branch-free
      kk0 += KSTEP;
      bool w1 = (kk0 >= D1); kk0 = w1 ? kk0 - D1 : kk0; jj += (w1 ? 1 : 0);
      bool w2 = (jj == D1);  jj  = w2 ? 0 : jj;         ii += (w2 ? 1 : 0);
    }
    __syncthreads();

    // ---- compute: wave w = N-tile w; 1 B fragment x 4 A fragments ----
    {
      const U256* bb = (const U256*)Blds;
      const U256* ab = (const U256*)Alds;
      v16bf b = __builtin_bit_cast(v16bf, bb[w * 32 + lane]);
      #pragma unroll
      for (int q = 0; q < 4; ++q) {
        v16bf a = __builtin_bit_cast(v16bf, ab[q * 32 + lane]);
        c[q] = __builtin_amdgcn_wmma_f32_16x16x32_bf16(
                   false, a, false, b, (short)0, c[q], false, false);
      }
    }
    __syncthreads();
  }

  // ---- tail: the single remaining K element kg = FUSED-1 = (64,64,64) ----
  if (kchunk == KCHUNKS - 1) {
    float bv = W1[(size_t)(w * 16 + m) * FUSED + (FUSED - 1)];
    #pragma unroll
    for (int q = 0; q < 4; ++q) {
      #pragma unroll
      for (int r = 0; r < 8; ++r) {
        int rl = q * 16 + r + hi * 8;                  // C-fragment row
        const float* R = &Xs[(rl * 3) * D1];
        c[q][r] += (R[D1 - 1] * R[2 * D1 - 1] * R[3 * D1 - 1]) * bv;
      }
    }
  }

  // ---- deterministic partial writeback: this kchunk's private slab ----
  float* slab = slabs + (size_t)kchunk * (BATCH * HID);
  const int col = w * 16 + m;
  #pragma unroll
  for (int q = 0; q < 4; ++q) {
    #pragma unroll
    for (int r = 0; r < 8; ++r) {
      int row = row0 + q * 16 + r + hi * 8;            // C-fragment row
      slab[row * HID + col] = c[q][r];
    }
  }
}

// ---------------- Kernel 1b: ordered slab reduction + bias + relu ----------
__global__ __launch_bounds__(256) void reduce_bias_relu_k1b(
    const float* __restrict__ slabs, const float* __restrict__ b1,
    float* __restrict__ y1)
{
  int u4 = (blockIdx.x * 256 + threadIdx.x) * 4;   // float4 per thread
  float4 acc = *(const float4*)&b1[u4 & (HID - 1)];
  #pragma unroll 4
  for (int kc = 0; kc < KCHUNKS; ++kc) {
    float4 s = *(const float4*)&slabs[(size_t)kc * (BATCH * HID) + u4];
    acc.x += s.x; acc.y += s.y; acc.z += s.z; acc.w += s.w;
  }
  float4 r;
  r.x = acc.x > 0.0f ? acc.x : 0.0f;
  r.y = acc.y > 0.0f ? acc.y : 0.0f;
  r.z = acc.z > 0.0f ? acc.z : 0.0f;
  r.w = acc.w > 0.0f ? acc.w : 0.0f;
  *(float4*)&y1[u4] = r;
}

// ---------------- Kernel 2: y2 = relu(y1 @ W2^T + b2) via WMMA -------------
__global__ __launch_bounds__(256) void mlp2_k2(
    const float* __restrict__ y1, const float* __restrict__ W2,
    const float* __restrict__ b2, float* __restrict__ y2)
{
  const int row0 = blockIdx.x * 16;              // 32 blocks cover 512 rows
  const int tid  = threadIdx.x;
  const int lane = tid & 31, w = tid >> 5;
  const int hi   = lane >> 4, m = lane & 15;
  const int h0   = w * 16;                       // wave's N tile

  v8f c = {};
  #pragma unroll
  for (int s = 0; s < 4; ++s) {                  // K = 128 in 4 steps
    const int k0 = s * KSTEP;
    // A: aligned float4 pairs from this lane's y1 row
    const float* arow = y1 + (size_t)(row0 + m) * HID + k0 + hi * 16;
    float4 a0 = *(const float4*)(arow + 0);
    float4 a1 = *(const float4*)(arow + 4);
    float4 a2 = *(const float4*)(arow + 8);
    float4 a3 = *(const float4*)(arow + 12);
    // B: 16 contiguous f32 from W2 row
    const float* wp = W2 + (size_t)(h0 + m) * HID + k0 + hi * 16;
    float4 b0 = *(const float4*)(wp + 0);
    float4 b1v = *(const float4*)(wp + 4);
    float4 b2v = *(const float4*)(wp + 8);
    float4 b3v = *(const float4*)(wp + 12);
    U256 au, bu;
    au.lo.x = pk_bf16x2(a0.x, a0.y); au.lo.y = pk_bf16x2(a0.z, a0.w);
    au.lo.z = pk_bf16x2(a1.x, a1.y); au.lo.w = pk_bf16x2(a1.z, a1.w);
    au.hi.x = pk_bf16x2(a2.x, a2.y); au.hi.y = pk_bf16x2(a2.z, a2.w);
    au.hi.z = pk_bf16x2(a3.x, a3.y); au.hi.w = pk_bf16x2(a3.z, a3.w);
    bu.lo.x = pk_bf16x2(b0.x, b0.y); bu.lo.y = pk_bf16x2(b0.z, b0.w);
    bu.lo.z = pk_bf16x2(b1v.x, b1v.y); bu.lo.w = pk_bf16x2(b1v.z, b1v.w);
    bu.hi.x = pk_bf16x2(b2v.x, b2v.y); bu.hi.y = pk_bf16x2(b2v.z, b2v.w);
    bu.hi.z = pk_bf16x2(b3v.x, b3v.y); bu.hi.w = pk_bf16x2(b3v.z, b3v.w);
    v16bf a = __builtin_bit_cast(v16bf, au);
    v16bf b = __builtin_bit_cast(v16bf, bu);
    c = __builtin_amdgcn_wmma_f32_16x16x32_bf16(
            false, a, false, b, (short)0, c, false, false);
  }
  const int col = h0 + m;
  #pragma unroll
  for (int r = 0; r < 8; ++r) {
    int row = row0 + r + hi * 8;
    float v = c[r] + b2[col];
    y2[row * HID + col] = v > 0.0f ? v : 0.0f;
  }
}

// ---------------- Kernel 3: head GEMV (N=8) + softmax, duplicated out ------
__global__ __launch_bounds__(256) void head_softmax_k3(
    const float* __restrict__ y2, const float* __restrict__ W3,
    const float* __restrict__ b3, float* __restrict__ out)
{
  int row = blockIdx.x * 256 + threadIdx.x;      // grid = 2 -> 512 rows
  float acc[OUT_N];
  #pragma unroll
  for (int o = 0; o < OUT_N; ++o) acc[o] = b3[o];
  const float* yr = y2 + (size_t)row * HID;
  for (int k = 0; k < HID; k += 4) {
    float4 yv = *(const float4*)(yr + k);
    #pragma unroll
    for (int o = 0; o < OUT_N; ++o) {
      const float* wr = W3 + o * HID + k;
      acc[o] += yv.x * wr[0] + yv.y * wr[1] + yv.z * wr[2] + yv.w * wr[3];
    }
  }
  float mx = acc[0];
  #pragma unroll
  for (int o = 1; o < OUT_N; ++o) mx = fmaxf(mx, acc[o]);
  float sum = 0.0f;
  #pragma unroll
  for (int o = 0; o < OUT_N; ++o) { acc[o] = __expf(acc[o] - mx); sum += acc[o]; }
  float inv = 1.0f / sum;
  #pragma unroll
  for (int o = 0; o < OUT_N; ++o) {
    float v = acc[o] * inv;
    out[row * OUT_N + o] = v;                    // first tuple element
    out[BATCH * OUT_N + row * OUT_N + o] = v;    // second tuple element
  }
}

// ---------------------------------------------------------------------------
extern "C" void kernel_launch(void* const* d_in, const int* in_sizes, int n_in,
                              void* d_out, int out_size, void* d_ws, size_t ws_size,
                              hipStream_t stream) {
  (void)in_sizes; (void)n_in; (void)out_size; (void)ws_size;
  const float* x  = (const float*)d_in[0];
  const float* W1 = (const float*)d_in[1];
  const float* b1 = (const float*)d_in[2];
  const float* W2 = (const float*)d_in[3];
  const float* b2 = (const float*)d_in[4];
  const float* W3 = (const float*)d_in[5];
  const float* b3 = (const float*)d_in[6];
  float* out = (float*)d_out;

  float* ws    = (float*)d_ws;
  float* slabs = ws;                                  // KCHUNKS * 512*128
  float* y1    = ws + (size_t)KCHUNKS * BATCH * HID;  // 512*128
  float* y2    = y1 + (size_t)BATCH * HID;            // 512*128

  fusion_gemm_k1<<<dim3(KCHUNKS, MBLOCKS), 256, 0, stream>>>(x, W1, slabs);
  reduce_bias_relu_k1b<<<(BATCH * HID) / (256 * 4), 256, 0, stream>>>(slabs, b1, y1);
  mlp2_k2<<<BATCH / 16, 256, 0, stream>>>(y1, W2, b2, y2);
  head_softmax_k3<<<BATCH / 256, 256, 0, stream>>>(y2, W3, b3, out);
}
